// Model_39676907885365
// MI455X (gfx1250) — matrix operand order repaired and checked by reading, not compiled, not run on hardware
//
#include <hip/hip_runtime.h>

// ---------------------------------------------------------------------------
// Flash-attention for: out = softmax((x @ y^T) * mask/sqrt(Skv)) @ y + x
// B=16, SQ=SKV=2048, D=256, fp32 in/out.  MI455X (gfx1250, wave32).
//
// v2 changes vs v1:
//  - Q fragments staged in LDS (frees 64 VGPRs -> aim below 256, no MSB
//    toggling, better occupancy for the HBM-bound mask stream)
//  - mask tiles streamed with global_load_async_to_lds_b128 (ASYNCcnt) into
//    a per-wave double buffer, one kv-step ahead; s_wait_asynccnt sync.
// ---------------------------------------------------------------------------

typedef __attribute__((ext_vector_type(16))) __bf16 v16bf;
typedef __attribute__((ext_vector_type(8)))  __bf16 v8bf;
typedef __attribute__((ext_vector_type(8)))  float  v8f;

#define NB   16
#define SQ   2048
#define SKV  2048
#define DD   256
#define XS   (NB * SQ * DD)
#define SCALE 0.022097086912079608f  // 1/sqrt(2048)

// ---------------------------------------------------------------------------
// Pre-pass: fp32 -> bf16 for x and y, plus per-batch transpose of y.
// ---------------------------------------------------------------------------
__global__ void __launch_bounds__(256) cvt_kernel(const float* __restrict__ x,
                                                  const float* __restrict__ y,
                                                  __bf16* __restrict__ xb,
                                                  __bf16* __restrict__ yb,
                                                  __bf16* __restrict__ yT) {
  unsigned i = blockIdx.x * 256u + threadIdx.x;
  float xv = x[i];
  float yv = y[i];
  xb[i] = (__bf16)xv;
  yb[i] = (__bf16)yv;
  unsigned d  = i & (DD - 1);
  unsigned bs = i / DD;
  unsigned s  = bs & (SKV - 1);
  unsigned b  = bs / SKV;
  yT[((size_t)b * DD + d) * SKV + s] = (__bf16)yv;
}

// ---------------------------------------------------------------------------
__device__ __forceinline__ float rmax16(float v) {
  v = fmaxf(v, __shfl_xor(v, 1, 16));
  v = fmaxf(v, __shfl_xor(v, 2, 16));
  v = fmaxf(v, __shfl_xor(v, 4, 16));
  v = fmaxf(v, __shfl_xor(v, 8, 16));
  return v;
}
__device__ __forceinline__ float rsum16(float v) {
  v += __shfl_xor(v, 1, 16);
  v += __shfl_xor(v, 2, 16);
  v += __shfl_xor(v, 4, 16);
  v += __shfl_xor(v, 8, 16);
  return v;
}

// LDS byte offset of a __shared__ object (flat addr low 32 bits = LDS addr).
__device__ __forceinline__ unsigned lds_off(const void* p) {
  return (unsigned)(uintptr_t)p;
}

// Async copy 16B from global to LDS (per active lane). Tracked by ASYNCcnt.
__device__ __forceinline__ void async_b128(unsigned lds_byte, const void* g) {
  asm volatile("global_load_async_to_lds_b128 %0, %1, off"
               :: "v"(lds_byte), "v"((unsigned long long)(uintptr_t)g)
               : "memory");
}
__device__ __forceinline__ void wait_async_le4() {
  asm volatile("s_wait_asynccnt 0x4" ::: "memory");
}
__device__ __forceinline__ void wait_async_0() {
  asm volatile("s_wait_asynccnt 0x0" ::: "memory");
}

// ---------------------------------------------------------------------------
// Main kernel: 256 threads = 8 waves; wave w owns query rows
// [qblk*128 + w*16, +16). kv streamed in tiles of 32, online softmax,
// all matrix math via v_wmma_f32_16x16x32_bf16.
// ---------------------------------------------------------------------------
__global__ void __launch_bounds__(256, 1)
attn_kernel(const float* __restrict__ x, const float* __restrict__ mask,
            const __bf16* __restrict__ xb, const __bf16* __restrict__ yb,
            const __bf16* __restrict__ yT, float* __restrict__ out) {
  // Q fragment staging: 8 waves x 8 frags x 32 lanes x 48B slot (32B used,
  // padded to 48B so b128 reads across lanes are bank-conflict-free). 96KB.
  __shared__ __attribute__((aligned(16))) char  qlds[8 * 8 * 32 * 48];
  // mask tiles: per-wave double buffer 16x32 f32, row padded to 36. 36KB.
  __shared__ __attribute__((aligned(16))) float mtile[8][2][16][36];
  // P staging (C-layout -> A-layout round trip), row padded to 40. 10KB.
  __shared__ __attribute__((aligned(32))) __bf16 plds[8][16][40];

  const int tid  = threadIdx.x;
  const int wave = tid >> 5;
  const int lane = tid & 31;
  const int m16  = lane & 15;
  const int half = lane >> 4;

  const int b     = blockIdx.x >> 4;   // 16 q-blocks per batch
  const int qblk  = blockIdx.x & 15;
  const int qbase = qblk * 128 + wave * 16;

  const __bf16* ybB = yb + ((size_t)b * SKV) * DD;
  const __bf16* yTB = yT + ((size_t)b * DD) * SKV;
  const float*  mB  = mask + ((size_t)b * SQ) * SKV;

  // ---- stage Q fragments into LDS (A-layout, 32B per lane per frag) ----
  {
    const __bf16* qrow = xb + ((size_t)(b * SQ + qbase + m16)) * DD;
#pragma unroll
    for (int kc = 0; kc < 8; ++kc) {
      v8bf lo = *(const v8bf*)(qrow + kc * 32 + half * 16);
      v8bf hi = *(const v8bf*)(qrow + kc * 32 + half * 16 + 8);
      v8bf* slot = (v8bf*)(qlds + ((size_t)((wave * 8 + kc) * 32 + lane)) * 48);
      slot[0] = lo;
      slot[1] = hi;
    }
  }

  // ---- issue async mask tile for kv0 = 0 (4 x b128 per lane = 2KB/wave) ----
  const int mrow = lane >> 1;        // 0..15
  const int mcol = (lane & 1) * 16;  // 0 or 16 (floats)
  const float* mrowp = mB + (size_t)(qbase + mrow) * SKV + mcol;
  {
#pragma unroll
    for (int i = 0; i < 4; ++i)
      async_b128(lds_off(&mtile[wave][0][mrow][mcol + i * 4]),
                 mrowp + i * 4);
  }

  // ---- accumulators ----
  v8f o[16];
  const v8f vzero = {};
#pragma unroll
  for (int nc = 0; nc < 16; ++nc) o[nc] = vzero;
  float mrowmax[8], lrow[8];
#pragma unroll
  for (int j = 0; j < 8; ++j) { mrowmax[j] = -__builtin_inff(); lrow[j] = 0.f; }

  int buf = 0;
  for (int kv0 = 0; kv0 < SKV; kv0 += 32) {
    // ---- prefetch next mask tile into other buffer, then wait current ----
    const int kvn = kv0 + 32;
    if (kvn < SKV) {
#pragma unroll
      for (int i = 0; i < 4; ++i)
        async_b128(lds_off(&mtile[wave][buf ^ 1][mrow][mcol + i * 4]),
                   mrowp + kvn + i * 4);
      wait_async_le4();
    } else {
      wait_async_0();
    }

    // ---------------- QK^T : S (16 x 32) ----------------
    v8f s0 = vzero, s1 = vzero;
    const __bf16* k0row = ybB + ((size_t)(kv0 + m16)) * DD + half * 16;
    const __bf16* k1row = ybB + ((size_t)(kv0 + 16 + m16)) * DD + half * 16;
#pragma unroll
    for (int kc = 0; kc < 8; ++kc) {
      const v8bf* slot =
          (const v8bf*)(qlds + ((size_t)((wave * 8 + kc) * 32 + lane)) * 48);
      v8bf qlo = slot[0];
      v8bf qhi = slot[1];
      v16bf qf;
#pragma unroll
      for (int i = 0; i < 8; ++i) { qf[i] = qlo[i]; qf[8 + i] = qhi[i]; }
      v16bf kf0 = *(const v16bf*)(k0row + kc * 32);
      s0 = __builtin_amdgcn_wmma_f32_16x16x32_bf16(false, qf, false, kf0,
                                                   (short)0, s0, false, false);
      v16bf kf1 = *(const v16bf*)(k1row + kc * 32);
      s1 = __builtin_amdgcn_wmma_f32_16x16x32_bf16(false, qf, false, kf1,
                                                   (short)0, s1, false, false);
    }

    // ------------- mask * scale, online softmax -------------
    float alpha[8];
#pragma unroll
    for (int j = 0; j < 8; ++j) {
      float mk0 = mtile[wave][buf][j + 8 * half][m16];
      float mk1 = mtile[wave][buf][j + 8 * half][16 + m16];
      float v0 = s0[j] * (mk0 * SCALE);
      float v1 = s1[j] * (mk1 * SCALE);
      float mx = rmax16(fmaxf(v0, v1));
      float mnew = fmaxf(mrowmax[j], mx);
      float al = __expf(mrowmax[j] - mnew);  // first iter: exp(-inf)=0
      mrowmax[j] = mnew;
      alpha[j] = al;
      float p0 = __expf(v0 - mnew);
      float p1 = __expf(v1 - mnew);
      lrow[j] = lrow[j] * al + rsum16(p0 + p1);
      plds[wave][j + 8 * half][m16]      = (__bf16)p0;
      plds[wave][j + 8 * half][16 + m16] = (__bf16)p1;
    }

    // rescale O accumulators (alpha[j] matches row j+8*half)
#pragma unroll
    for (int nc = 0; nc < 16; ++nc) {
#pragma unroll
      for (int j = 0; j < 8; ++j) o[nc][j] *= alpha[j];
    }

    // re-load P in A-layout (same-wave LDS ops are in-order; no barrier)
    const v8bf pa = *(const v8bf*)&plds[wave][m16][half * 16];
    const v8bf pb = *(const v8bf*)&plds[wave][m16][half * 16 + 8];
    v16bf pf;
#pragma unroll
    for (int i = 0; i < 8; ++i) { pf[i] = pa[i]; pf[8 + i] = pb[i]; }

    // ---------------- O += P @ V ----------------
#pragma unroll
    for (int nc = 0; nc < 16; ++nc) {
      const __bf16* vp =
          yTB + ((size_t)(nc * 16 + m16)) * SKV + kv0 + half * 16;
      v16bf vf = *(const v16bf*)vp;
      o[nc] = __builtin_amdgcn_wmma_f32_16x16x32_bf16(false, pf, false, vf,
                                                      (short)0, o[nc], false,
                                                      false);
    }

    buf ^= 1;
  }

  // ---------------- epilogue: normalize, add residual x ----------------
#pragma unroll
  for (int nc = 0; nc < 16; ++nc) {
#pragma unroll
    for (int j = 0; j < 8; ++j) {
      const int row = qbase + j + 8 * half;
      const int col = nc * 16 + m16;
      const size_t idx = ((size_t)(b * SQ + row)) * DD + col;
      out[idx] = o[nc][j] / lrow[j] + x[idx];
    }
  }
}

// ---------------------------------------------------------------------------
extern "C" void kernel_launch(void* const* d_in, const int* in_sizes, int n_in,
                              void* d_out, int out_size, void* d_ws,
                              size_t ws_size, hipStream_t stream) {
  const float* x    = (const float*)d_in[0];
  const float* y    = (const float*)d_in[1];
  const float* mask = (const float*)d_in[2];
  // d_in[3] = dropout scalar (p=0) -> ignored

  __bf16* xb = (__bf16*)d_ws;
  __bf16* yb = xb + XS;
  __bf16* yT = yb + XS;   // total 48 MB workspace

  cvt_kernel<<<XS / 256, 256, 0, stream>>>(x, y, xb, yb, yT);

  attn_kernel<<<NB * (SQ / 128), 256, 0, stream>>>(
      x, mask, xb, yb, yT, (float*)d_out);
}
